// AttentionBlock_51049981281188
// MI455X (gfx1250) — matrix operand order repaired and checked by reading, not compiled, not run on hardware
//
#include <hip/hip_runtime.h>
#include <math.h>

// ---------------------------------------------------------------------------
// Problem constants (B,C,H,W = 8,512,64,64 ; GROUPS=32)
// ---------------------------------------------------------------------------
#define NBATCH 8
#define CDIM   512
#define NPIX   4096            // H*W
#define NGROUP 32
#define CPG    16              // CDIM / NGROUP
#define GNELEM (CPG * NPIX)    // 65536 elements per (batch, group)
#define SCALE_QK 0.21022410381342863f   // 512^-0.25
#define GN_EPS 1e-5f

typedef __attribute__((ext_vector_type(16))) __bf16 v16bf;
typedef __attribute__((ext_vector_type(8)))  __bf16 v8bf;
typedef __attribute__((ext_vector_type(8)))  float  v8f;

static __device__ __forceinline__ v8f wmma_bf16(v16bf a, v16bf b, v8f c) {
    // D = A(16x32) * B(32x16) + C, f32 accumulate
    return __builtin_amdgcn_wmma_f32_16x16x32_bf16(
        false, a, false, b, (short)0, c, false, false);
}

// A-fragment (16x32 bf16, M x K). Memory layout: row m contiguous in k,
// row stride = `stride` elements.  Order used here: lane(m=lane&15, half=lane>>4)
// holds k = e + half*16 (matches B fragment order).
static __device__ __forceinline__ v16bf load_a_frag(const __bf16* base, int stride, int lane) {
    const int m = lane & 15, hlf = lane >> 4;
    const __bf16* p = base + (size_t)m * stride + hlf * 16;
    v8bf lo = *(const v8bf*)(p);
    v8bf hi = *(const v8bf*)(p + 8);
    return __builtin_shufflevector(lo, hi, 0,1,2,3,4,5,6,7,8,9,10,11,12,13,14,15);
}

// B-fragment (32x16 bf16, K x N). Memory layout: column n is a contiguous
// run of 32 k-values, column stride = `stride`.  Order used here: lane(n=lane&15,
// half=lane>>4) holds k = e + half*16  -> one contiguous 32B run per lane.
static __device__ __forceinline__ v16bf load_b_frag(const __bf16* base, int stride, int lane) {
    const int n = lane & 15, hlf = lane >> 4;
    return *(const v16bf*)(base + (size_t)n * stride + hlf * 16);
}

// ---------------------------------------------------------------------------
// Kernel 1: GroupNorm, output transposed bf16:  xnT[b][n][c]
// ---------------------------------------------------------------------------
__global__ void gn_kernel(const float* __restrict__ x,
                          const float* __restrict__ gamma,
                          const float* __restrict__ beta,
                          __bf16* __restrict__ xnT) {
    const int bg = blockIdx.x;              // b*NGROUP + g
    const int b  = bg >> 5, g = bg & 31;
    const size_t xbase = (size_t)b * CDIM * NPIX + (size_t)g * CPG * NPIX;
    const int tid = threadIdx.x;

    float s = 0.f, ss = 0.f;
    for (int i = tid; i < GNELEM; i += 256) {
        float v = x[xbase + i];
        s += v; ss += v * v;
    }
    __shared__ float rs[256], rss[256];
    rs[tid] = s; rss[tid] = ss;
    __syncthreads();
    for (int off = 128; off > 0; off >>= 1) {
        if (tid < off) { rs[tid] += rs[tid + off]; rss[tid] += rss[tid + off]; }
        __syncthreads();
    }
    const float mu  = rs[0] * (1.f / GNELEM);
    const float var = rss[0] * (1.f / GNELEM) - mu * mu;
    const float inv = rsqrtf(var + GN_EPS);

    __bf16* dst = xnT + (size_t)b * NPIX * CDIM;
    for (int i = tid; i < GNELEM; i += 256) {
        int cl = i >> 12;            // channel within group
        int n  = i & (NPIX - 1);
        int c  = g * CPG + cl;
        float v = (x[xbase + i] - mu) * inv * gamma[c] + beta[c];
        dst[(size_t)n * CDIM + c] = (__bf16)v;
    }
}

// ---------------------------------------------------------------------------
// Kernel 2: convert weights f32 -> bf16 (done once, reused by every tile)
// ---------------------------------------------------------------------------
__global__ void convert_w_kernel(const float* __restrict__ qkv_w,
                                 const float* __restrict__ proj_w,
                                 __bf16* __restrict__ wq,
                                 __bf16* __restrict__ wp) {
    int i = blockIdx.x * blockDim.x + threadIdx.x;
    const int NQ = 3 * CDIM * CDIM;          // 786432
    const int NP = CDIM * CDIM;              // 262144
    if (i < NQ)            wq[i] = (__bf16)qkv_w[i];
    else if (i < NQ + NP)  wp[i - NQ] = (__bf16)proj_w[i - NQ];
}

// ---------------------------------------------------------------------------
// Kernel 3: QKV 1x1-conv GEMM:  qkv[o,n] = sum_c w[o,c] * xnT[n,c]
//   outputs: qT[b][n][c] (scaled), kT[b][n][c] (scaled), v[b][c][n]
//   block = 128 threads (4 waves); block tile = 64(o) x 64(n)
//   Software-pipelined: fragments for step kc+1 are loaded before the
//   WMMAs of step kc so load latency overlaps matrix ops.
// ---------------------------------------------------------------------------
__global__ void qkv_kernel(const __bf16* __restrict__ xnT,
                           const __bf16* __restrict__ wq,
                           const float*  __restrict__ qkv_b,
                           __bf16* __restrict__ qT,
                           __bf16* __restrict__ kT,
                           __bf16* __restrict__ vM) {
    const int lane = threadIdx.x & 31, wave = threadIdx.x >> 5;
    const int n0 = blockIdx.x * 64;
    const int o0 = blockIdx.y * 64 + wave * 16;
    const int b  = blockIdx.z;
    const __bf16* xb = xnT + (size_t)b * NPIX * CDIM;
    const __bf16* wbase = wq + (size_t)o0 * CDIM;

    v8f acc[4] = {};
    v16bf a_cur = load_a_frag(wbase, CDIM, lane);
    v16bf b_cur[4];
#pragma unroll
    for (int nt = 0; nt < 4; ++nt)
        b_cur[nt] = load_b_frag(xb + (size_t)(n0 + nt * 16) * CDIM, CDIM, lane);

#pragma unroll
    for (int kc = 0; kc < CDIM / 32; ++kc) {
        v16bf a_nxt;
        v16bf b_nxt[4];
        if (kc + 1 < CDIM / 32) {
            a_nxt = load_a_frag(wbase + (kc + 1) * 32, CDIM, lane);
#pragma unroll
            for (int nt = 0; nt < 4; ++nt)
                b_nxt[nt] = load_b_frag(xb + (size_t)(n0 + nt * 16) * CDIM + (kc + 1) * 32,
                                        CDIM, lane);
        }
#pragma unroll
        for (int nt = 0; nt < 4; ++nt)
            acc[nt] = wmma_bf16(a_cur, b_cur[nt], acc[nt]);
        a_cur = a_nxt;
#pragma unroll
        for (int nt = 0; nt < 4; ++nt) b_cur[nt] = b_nxt[nt];
    }

    const int hlf = lane >> 4, nl = lane & 15;
    float bias[8];
#pragma unroll
    for (int r = 0; r < 8; ++r) bias[r] = qkv_b[o0 + hlf * 8 + r];

    if (o0 < 2 * CDIM) {                     // q or k -> transposed, scaled
        __bf16* dst = (o0 < CDIM ? qT : kT) + (size_t)b * NPIX * CDIM;
        const int ob = (o0 < CDIM) ? o0 : o0 - CDIM;
#pragma unroll
        for (int nt = 0; nt < 4; ++nt) {
            int n = n0 + nt * 16 + nl;
            v8bf pk;
#pragma unroll
            for (int r = 0; r < 8; ++r)
                pk[r] = (__bf16)((acc[nt][r] + bias[r]) * SCALE_QK);
            *(v8bf*)(dst + (size_t)n * CDIM + ob + hlf * 8) = pk;
        }
    } else {                                 // v -> [c][n] layout
        __bf16* dst = vM + (size_t)b * CDIM * NPIX;
        const int ob = o0 - 2 * CDIM;
#pragma unroll
        for (int nt = 0; nt < 4; ++nt) {
            int n = n0 + nt * 16 + nl;
#pragma unroll
            for (int r = 0; r < 8; ++r)
                dst[(size_t)(ob + hlf * 8 + r) * NPIX + n] = (__bf16)(acc[nt][r] + bias[r]);
        }
    }
}

// ---------------------------------------------------------------------------
// Kernel 4: flash attention.
//   block = 128 threads (4 waves) handles 16 query rows, all 512 V channels.
//   wave w: computes S for keys [w*16, w*16+16) of each 64-key tile and
//   owns V channels [w*128, (w+1)*128) of the output accumulator.
// ---------------------------------------------------------------------------
__global__ void attn_kernel(const __bf16* __restrict__ qT,
                            const __bf16* __restrict__ kT,
                            const __bf16* __restrict__ vM,
                            __bf16* __restrict__ hT) {
    const int lane = threadIdx.x & 31, wave = threadIdx.x >> 5;
    const int n0 = blockIdx.x * 16;          // query rows
    const int b  = blockIdx.y;
    const __bf16* qb = qT + (size_t)b * NPIX * CDIM;
    const __bf16* kb = kT + (size_t)b * NPIX * CDIM;
    const __bf16* vb = vM + (size_t)b * CDIM * NPIX;

    // Register-resident Q fragments (16 rows x 512 channels)
    v16bf qf[16];
#pragma unroll
    for (int kc = 0; kc < 16; ++kc)
        qf[kc] = load_a_frag(qb + (size_t)n0 * CDIM + kc * 32, CDIM, lane);

    v8f acc[8] = {};                         // 16 q-rows x 128 channels (this wave)
    const int c0 = wave * 128;

    // Online-softmax state: thread tid handles row tid/8, 8 threads per row.
    const int srow = threadIdx.x >> 3;
    const int sj   = threadIdx.x & 7;
    float m_i = -INFINITY, l_i = 0.f;

    __shared__ float  Sf[16][68];
    __shared__ __bf16 Pbf[16][64];
    __shared__ float  facArr[16];
    __shared__ float  lArr[16];

    const int hlf = lane >> 4, nl = lane & 15;

    for (int t = 0; t < NPIX / 64; ++t) {
        const int m0 = t * 64;

        // Prefetch next key tile (contiguous 64 rows * 1KB) toward this WGP.
        if (t + 1 < NPIX / 64) {
            const __bf16* nk = kb + (size_t)(m0 + 64) * CDIM;
            __builtin_prefetch(nk + (size_t)threadIdx.x * 256, 0, 0);
        }

        // ---- S = qT * k   (16 x 16 per wave, contraction over C=512) ----
        v8f s = {};
        const __bf16* kbase = kb + (size_t)(m0 + wave * 16) * CDIM;
        {
            v16bf bf_cur = load_b_frag(kbase, CDIM, lane);
#pragma unroll
            for (int kc = 0; kc < 16; ++kc) {
                v16bf bf_nxt;
                if (kc + 1 < 16)
                    bf_nxt = load_b_frag(kbase + (kc + 1) * 32, CDIM, lane);
                s = wmma_bf16(qf[kc], bf_cur, s);
                bf_cur = bf_nxt;
            }
        }
        {
            const int col = wave * 16 + nl;
#pragma unroll
            for (int r = 0; r < 8; ++r) Sf[r + hlf * 8][col] = s[r];
        }
        __syncthreads();

        // ---- online softmax over the 64-key tile ----
        {
            float vals[8];
            float mx = -INFINITY;
#pragma unroll
            for (int j = 0; j < 8; ++j) {
                vals[j] = Sf[srow][sj * 8 + j];
                mx = fmaxf(mx, vals[j]);
            }
            for (int off = 1; off < 8; off <<= 1)
                mx = fmaxf(mx, __shfl_xor(mx, off, 32));
            const float m_new = fmaxf(m_i, mx);
            const float fac = expf(m_i - m_new);
            float sum = 0.f;
#pragma unroll
            for (int j = 0; j < 8; ++j) {
                float p = expf(vals[j] - m_new);
                sum += p;
                Pbf[srow][sj * 8 + j] = (__bf16)p;
            }
            for (int off = 1; off < 8; off <<= 1)
                sum += __shfl_xor(sum, off, 32);
            l_i = l_i * fac + sum;
            m_i = m_new;
            if (sj == 0) facArr[srow] = fac;
        }
        __syncthreads();

        // ---- rescale accumulators ----
        {
            float fr[8];
#pragma unroll
            for (int r = 0; r < 8; ++r) fr[r] = facArr[r + hlf * 8];
#pragma unroll
            for (int nf = 0; nf < 8; ++nf)
#pragma unroll
                for (int r = 0; r < 8; ++r) acc[nf][r] *= fr[r];
        }

        // ---- O += P(16x64) * V^T(64x128 per wave), pipelined over nf ----
#pragma unroll
        for (int kk = 0; kk < 2; ++kk) {
            v16bf pf = load_a_frag(&Pbf[0][0] + kk * 32, 64, lane);
            v16bf bf_cur = load_b_frag(vb + (size_t)c0 * NPIX + m0 + kk * 32, NPIX, lane);
#pragma unroll
            for (int nf = 0; nf < 8; ++nf) {
                v16bf bf_nxt;
                if (nf + 1 < 8)
                    bf_nxt = load_b_frag(vb + (size_t)(c0 + (nf + 1) * 16) * NPIX + m0 + kk * 32,
                                         NPIX, lane);
                acc[nf] = wmma_bf16(pf, bf_cur, acc[nf]);
                bf_cur = bf_nxt;
            }
        }
        __syncthreads();
    }

    // ---- finalize: divide by row sums, store hT[b][n][c] (bf16) ----
    if (sj == 0) lArr[srow] = l_i;
    __syncthreads();
    float inv[8];
#pragma unroll
    for (int r = 0; r < 8; ++r) inv[r] = 1.f / lArr[r + hlf * 8];

    __bf16* hb = hT + (size_t)b * NPIX * CDIM;
#pragma unroll
    for (int nf = 0; nf < 8; ++nf) {
        int c = c0 + nf * 16 + nl;
#pragma unroll
        for (int r = 0; r < 8; ++r)
            hb[(size_t)(n0 + r + hlf * 8) * CDIM + c] = (__bf16)(acc[nf][r] * inv[r]);
    }
}

// ---------------------------------------------------------------------------
// Kernel 5: output projection + bias + residual:
//   out[b,o,n] = x[b,o,n] + proj_b[o] + sum_c pw[o,c] * hT[b,n,c]
// ---------------------------------------------------------------------------
__global__ void proj_kernel(const __bf16* __restrict__ hT,
                            const __bf16* __restrict__ wp,
                            const float*  __restrict__ proj_b,
                            const float*  __restrict__ x,
                            float* __restrict__ out) {
    const int lane = threadIdx.x & 31, wave = threadIdx.x >> 5;
    const int n0 = blockIdx.x * 64;
    const int o0 = blockIdx.y * 64 + wave * 16;
    const int b  = blockIdx.z;
    const __bf16* hb = hT + (size_t)b * NPIX * CDIM;
    const __bf16* wbase = wp + (size_t)o0 * CDIM;

    v8f acc[4] = {};
    v16bf a_cur = load_a_frag(wbase, CDIM, lane);
    v16bf b_cur[4];
#pragma unroll
    for (int nt = 0; nt < 4; ++nt)
        b_cur[nt] = load_b_frag(hb + (size_t)(n0 + nt * 16) * CDIM, CDIM, lane);

#pragma unroll
    for (int kc = 0; kc < CDIM / 32; ++kc) {
        v16bf a_nxt;
        v16bf b_nxt[4];
        if (kc + 1 < CDIM / 32) {
            a_nxt = load_a_frag(wbase + (kc + 1) * 32, CDIM, lane);
#pragma unroll
            for (int nt = 0; nt < 4; ++nt)
                b_nxt[nt] = load_b_frag(hb + (size_t)(n0 + nt * 16) * CDIM + (kc + 1) * 32,
                                        CDIM, lane);
        }
#pragma unroll
        for (int nt = 0; nt < 4; ++nt)
            acc[nt] = wmma_bf16(a_cur, b_cur[nt], acc[nt]);
        a_cur = a_nxt;
#pragma unroll
        for (int nt = 0; nt < 4; ++nt) b_cur[nt] = b_nxt[nt];
    }

    const int hlf = lane >> 4, nl = lane & 15;
    float bias[8];
#pragma unroll
    for (int r = 0; r < 8; ++r) bias[r] = proj_b[o0 + hlf * 8 + r];

#pragma unroll
    for (int nt = 0; nt < 4; ++nt) {
        int n = n0 + nt * 16 + nl;
#pragma unroll
        for (int r = 0; r < 8; ++r) {
            int o = o0 + hlf * 8 + r;
            size_t idx = (size_t)b * CDIM * NPIX + (size_t)o * NPIX + n;
            out[idx] = x[idx] + acc[nt][r] + bias[r];
        }
    }
}

// ---------------------------------------------------------------------------
// Launch
// ---------------------------------------------------------------------------
extern "C" void kernel_launch(void* const* d_in, const int* in_sizes, int n_in,
                              void* d_out, int out_size, void* d_ws, size_t ws_size,
                              hipStream_t stream) {
    const float* x      = (const float*)d_in[0];
    const float* gamma  = (const float*)d_in[1];
    const float* beta   = (const float*)d_in[2];
    const float* qkv_w  = (const float*)d_in[3];
    const float* qkv_b  = (const float*)d_in[4];
    const float* proj_w = (const float*)d_in[5];
    const float* proj_b = (const float*)d_in[6];
    float* out = (float*)d_out;

    // workspace layout (bf16 elements); each activation buffer = B*N*C
    const size_t ACT = (size_t)NBATCH * NPIX * CDIM;   // 16,777,216 elems
    __bf16* base = (__bf16*)d_ws;
    __bf16* xnT = base;
    __bf16* qT  = base + 1 * ACT;
    __bf16* kT  = base + 2 * ACT;
    __bf16* vM  = base + 3 * ACT;
    __bf16* hT  = base + 4 * ACT;
    __bf16* wqB = base + 5 * ACT;                      // 786432 elems
    __bf16* wpB = wqB + (size_t)3 * CDIM * CDIM;       // 262144 elems

    // 1. GroupNorm -> xnT (bf16, transposed)
    gn_kernel<<<dim3(NBATCH * NGROUP), dim3(256), 0, stream>>>(x, gamma, beta, xnT);

    // 2. weights -> bf16
    convert_w_kernel<<<dim3((3 * CDIM * CDIM + CDIM * CDIM + 255) / 256), dim3(256), 0, stream>>>(
        qkv_w, proj_w, wqB, wpB);

    // 3. QKV projection (WMMA) -> qT, kT, v
    qkv_kernel<<<dim3(NPIX / 64, 3 * CDIM / 64, NBATCH), dim3(128), 0, stream>>>(
        xnT, wqB, qkv_b, qT, kT, vM);

    // 4. flash attention (WMMA) -> hT
    attn_kernel<<<dim3(NPIX / 16, NBATCH), dim3(128), 0, stream>>>(qT, kT, vM, hT);

    // 5. output projection + residual (WMMA) -> out
    proj_kernel<<<dim3(NPIX / 64, CDIM / 64, NBATCH), dim3(128), 0, stream>>>(
        hT, wpB, proj_b, x, out);
}
